// CrossTransformer_37520834298096
// MI455X (gfx1250) — matrix operand order repaired and checked by reading, not compiled, not run on hardware
//
#include <hip/hip_runtime.h>
#include <hip/hip_bf16.h>

// ---------------------------------------------------------------------------
// CrossTransformer for MI455X (gfx1250, wave32, WMMA + TDM).
// Pass 1: bf16 projection GEMMs (w_qk/w_v x image) with LDS staging.
// Pass 2: flash-attention per (b,n); K/V blocks staged into LDS by the
//         Tensor Data Mover (tensor_load_to_lds, TENSORcnt), double-buffered,
//         with DMA-inserted row padding so ds_load_b128 B-fragments are
//         bank-conflict-free. Squared-distance epilogue with f32 atomics.
// Precision: bf16 WMMA (v_wmma_f32_16x16x32_bf16), f32 accumulate/softmax.
// ---------------------------------------------------------------------------

typedef __bf16 bf16;
typedef __attribute__((ext_vector_type(4)))  bf16  bf16x4;
typedef __attribute__((ext_vector_type(8)))  bf16  bf16x8;
typedef __attribute__((ext_vector_type(16))) bf16  bf16x16;
typedef __attribute__((ext_vector_type(8)))  float f32x8;
typedef __attribute__((ext_vector_type(4)))  float f32x4;
typedef unsigned int u32x4 __attribute__((ext_vector_type(4)));
typedef int          i32x4 __attribute__((ext_vector_type(4)));
typedef int          i32x8 __attribute__((ext_vector_type(8)));

#define WMMA_BF16(a, b, c) \
  __builtin_amdgcn_wmma_f32_16x16x32_bf16(false, (a), false, (b), (short)0, (c), false, false)

#define HAVE_TDM __has_builtin(__builtin_amdgcn_tensor_load_to_lds)

static __device__ inline bf16x16 join16(bf16x8 lo, bf16x8 hi) {
  return __builtin_shufflevector(lo, hi, 0,1,2,3,4,5,6,7,8,9,10,11,12,13,14,15);
}
static __device__ inline f32x8 zero8() {
  f32x8 z; for (int e = 0; e < 8; ++e) z[e] = 0.0f; return z;
}

#if HAVE_TDM
// Issue a 2D TDM tile load: tile_d1 rows of tile_d0 bf16 elements, global row
// stride `stride0` elements, destination LDS at byte offset `lds_off`.
// pad_enable with pad_amount=16B per pad_interval (one row) de-conflicts the
// LDS banks for the later ds_load_b128 fragment reads.
static __device__ inline void tdm_load_2d(unsigned lds_off, const void* gptr,
                                          unsigned tile_d0, unsigned tile_d1,
                                          unsigned tensor_d0, unsigned tensor_d1,
                                          unsigned stride0, unsigned itv_code)
{
  unsigned long long ga = (unsigned long long)(uintptr_t)gptr;
  u32x4 g0;
  g0[0] = 1u;                                        // count=1 (valid), user mode
  g0[1] = lds_off;                                   // [63:32]  lds_addr (bytes)
  g0[2] = (unsigned)(ga & 0xffffffffu);              // [95:64]  global_addr lo
  g0[3] = (unsigned)((ga >> 32) & 0x01ffffffu)       // [120:96] global_addr hi
        | (2u << 30);                                // [127:126] type=2 (image)
  i32x8 g1;
  g1[0] = (int)((1u << 16)                           // data_size = 2 bytes
              | (1u << 20)                           // pad_enable
              | (itv_code << 22)                     // pad_interval (one row)
              | (3u << 25));                         // pad_amount = 4 dwords (16B)
  g1[1] = (int)((tensor_d0 & 0xffffu) << 16);        // tensor_dim0 [63:48]
  g1[2] = (int)((tensor_d0 >> 16) | ((tensor_d1 & 0xffffu) << 16));
  g1[3] = (int)((tensor_d1 >> 16) | (tile_d0 << 16));// tile_dim0 [127:112]
  g1[4] = (int)tile_d1;                              // tile_dim1; tile_dim2=0
  g1[5] = (int)stride0;                              // tensor_dim0_stride lo
  g1[6] = 0;
  g1[7] = 0;
  i32x4 z4; z4[0] = z4[1] = z4[2] = z4[3] = 0;
#if defined(__clang_major__) && __clang_major__ >= 23
  i32x8 z8; for (int i = 0; i < 8; ++i) z8[i] = 0;
  __builtin_amdgcn_tensor_load_to_lds(g0, g1, z4, z4, z8, 0);
#else
  __builtin_amdgcn_tensor_load_to_lds(g0, g1, z4, z4, 0);
#endif
}
#endif  // HAVE_TDM

// ---------------------------------------------------------------------------
// Kernel 1: projections.
// Grid: 832 = 416 images x 2 hw-halves. Block: 256 threads = 8 waves.
// Each WG: D[256 x 128] = [Wqk;Wv][256 x 512] x X[512 x 128cols], staged in
// LDS as bf16 (A-layout rows, B-layout transposed X), 16 k-steps of 32.
// Outputs:
//   rows   0..127 (K-proj): Qt/Kt [pixel][dk] bf16   (16B vector stores)
//   rows 128..255 (V-proj): query -> QV [hw][dv] f32; support -> Vn [dv][key]
// ---------------------------------------------------------------------------
__global__ void __launch_bounds__(256)
proj_kernel(const float* __restrict__ qrep, const float* __restrict__ srep,
            const float* __restrict__ wqk,  const float* __restrict__ wv,
            bf16* __restrict__ Qt, float* __restrict__ QV,
            bf16* __restrict__ Kt, bf16* __restrict__ Vn)
{
  __shared__ bf16 Ws[256 * 40];   // W slice  [256 rows][32 k] (+8 pad)
  __shared__ bf16 Xs[128 * 40];   // X slice  [128 hw ][32 k] (+8 pad), transposed

  const int tid  = threadIdx.x;
  const int wave = tid >> 5;
  const int lane = tid & 31;
  const int ln   = lane & 15;
  const int hh   = lane >> 4;

  const int img   = blockIdx.x >> 1;
  const int ncol0 = (blockIdx.x & 1) * 128;
  const bool is_query = (img < 16);
  const float* X = is_query ? (qrep + (size_t)img * 512 * 256)
                            : (srep + (size_t)(img - 16) * 512 * 256);

  f32x8 acc[2][8];
  for (int i = 0; i < 2; ++i)
    for (int j = 0; j < 8; ++j) acc[i][j] = zero8();

  for (int ks = 0; ks < 16; ++ks) {
    const int k0 = ks * 32;
    for (int i = 0; i < 8; ++i) {                    // stage W slice
      int lin = (i * 256 + tid) * 4;
      int row = lin >> 5, kc = lin & 31;
      const float* src = (row < 128) ? (wqk + row * 512 + k0 + kc)
                                     : (wv + (row - 128) * 512 + k0 + kc);
      f32x4 v = *(const f32x4*)src;
      bf16x4 pv;
      pv[0] = (bf16)v[0]; pv[1] = (bf16)v[1]; pv[2] = (bf16)v[2]; pv[3] = (bf16)v[3];
      *(bf16x4*)(Ws + row * 40 + kc) = pv;
    }
    for (int i = 0; i < 4; ++i) {                    // stage X slice transposed
      int lin = (i * 256 + tid) * 4;
      int c = lin >> 7, hw = lin & 127;
      f32x4 v = *(const f32x4*)(X + (size_t)(k0 + c) * 256 + ncol0 + hw);
      Xs[(hw + 0) * 40 + c] = (bf16)v[0];
      Xs[(hw + 1) * 40 + c] = (bf16)v[1];
      Xs[(hw + 2) * 40 + c] = (bf16)v[2];
      Xs[(hw + 3) * 40 + c] = (bf16)v[3];
    }
    __syncthreads();

    bf16x16 afr[2];
    for (int mt = 0; mt < 2; ++mt) {                 // A fragments (ISA layout)
      const bf16* p = Ws + (wave * 32 + mt * 16 + ln) * 40 + hh * 8;
      afr[mt] = join16(*(const bf16x8*)p, *(const bf16x8*)(p + 16));
    }
    for (int nt = 0; nt < 8; ++nt) {                 // B fragments + WMMA
      const bf16* p = Xs + (nt * 16 + ln) * 40 + hh * 8;
      bf16x16 bfr = join16(*(const bf16x8*)p, *(const bf16x8*)(p + 16));
      acc[0][nt] = WMMA_BF16(afr[0], bfr, acc[0][nt]);
      acc[1][nt] = WMMA_BF16(afr[1], bfr, acc[1][nt]);
    }
    __syncthreads();
  }

  if (wave < 4) {                                    // K-proj -> [pixel][dk]
    bf16* base = is_query ? (Qt + (size_t)img * 256 * 128)
                          : (Kt + (size_t)(img - 16) * 256 * 128);
    for (int mt = 0; mt < 2; ++mt)
      for (int nt = 0; nt < 8; ++nt) {
        int dk0 = wave * 32 + mt * 16 + hh * 8;
        int pix = ncol0 + nt * 16 + ln;
        bf16x8 p;
        for (int r = 0; r < 8; ++r) p[r] = (bf16)acc[mt][nt][r];
        *(bf16x8*)(base + (size_t)pix * 128 + dk0) = p;
      }
  } else if (is_query) {                             // query V -> QV f32 [hw][dv]
    float* base = QV + (size_t)img * 256 * 128;
    for (int mt = 0; mt < 2; ++mt)
      for (int nt = 0; nt < 8; ++nt) {
        int dv0 = (wave - 4) * 32 + mt * 16 + hh * 8;
        int pix = ncol0 + nt * 16 + ln;
        f32x4 lo, hi;
        for (int r = 0; r < 4; ++r) { lo[r] = acc[mt][nt][r]; hi[r] = acc[mt][nt][r + 4]; }
        *(f32x4*)(base + (size_t)pix * 128 + dv0)     = lo;
        *(f32x4*)(base + (size_t)pix * 128 + dv0 + 4) = hi;
      }
  } else {                                           // support V -> Vn [dv][key]
    int sidx = img - 16;
    bf16* base = Vn + (size_t)(sidx / 5) * 128 * 1280 + (sidx % 5) * 256;
    for (int mt = 0; mt < 2; ++mt)
      for (int nt = 0; nt < 8; ++nt) {
        int pix = ncol0 + nt * 16 + ln;
        for (int r = 0; r < 8; ++r) {
          int dv = (wave - 4) * 32 + mt * 16 + hh * 8 + r;
          base[(size_t)dv * 1280 + pix] = (bf16)acc[mt][nt][r];
        }
      }
  }
}

// ---------------------------------------------------------------------------
// Kernel 2: flash attention + distance.
// Grid: 320 = (b*5+n) x 4 query-row blocks of 64. Block: 128 = 4 waves.
// Per key-block of 64 (20 blocks): K[64x128] and V[128x64] bf16 tiles are
// DMA'd into LDS by the TDM (wave 0 issues, double-buffered, prefetching
// block kb+1 during compute of kb; s_wait_tensorcnt + barrier for handoff).
// DMA inserts 16B row padding: K row stride 272B, V row stride 144B -> all
// 16 lanes of a B-fragment hit distinct LDS banks.
//   S[16x64] = Q[16x128] K^T     (16 WMMA / wave)
//   online softmax (shfl_xor within 16-lane halves)
//   P -> LDS (D-layout -> A-layout), s_wait_dscnt
//   O[16x128] += P[16x64] V      (16 WMMA / wave)
// ---------------------------------------------------------------------------
__global__ void __launch_bounds__(128)
attn_kernel(const bf16* __restrict__ Qt, const float* __restrict__ QV,
            const bf16* __restrict__ Kt, const bf16* __restrict__ Vn,
            float* __restrict__ out)
{
  __shared__ __align__(16) bf16 Kls[2][64 * 136];   // [key][128+8] halfs, 272B rows
  __shared__ __align__(16) bf16 Vls[2][128 * 72];   // [dv ][ 64+8] halfs, 144B rows
  __shared__ __align__(16) bf16 Pls[4 * 16 * 80];   // per-wave P [16][64+16]

  const int tid  = threadIdx.x;
  const int wave = tid >> 5;
  const int lane = tid & 31;
  const int ln   = lane & 15;
  const int hh   = lane >> 4;

  const int bn    = blockIdx.x >> 2;
  const int qb    = blockIdx.x & 3;
  const int b     = bn / 5;
  const int qrow0 = qb * 64 + wave * 16;

  const bf16* Qb = Qt + (size_t)b  * 256 * 128;
  const bf16* Kb = Kt + (size_t)bn * 1280 * 128;
  const bf16* Vb = Vn + (size_t)bn * 128 * 1280;
  bf16* Pw = Pls + wave * 16 * 80;

#if HAVE_TDM
  auto stage = [&](int kb2, int buf) {
    // K block: 64 rows x 128 elems, row stride 128; pad every 256B (itv=5)
    tdm_load_2d((unsigned)(uintptr_t)&Kls[buf][0],
                Kb + (size_t)(kb2 * 64) * 128,
                128, 64, 128, 1280, 128, 5u);
    // V block: 128 rows x 64 elems, row stride 1280; pad every 128B (itv=4)
    tdm_load_2d((unsigned)(uintptr_t)&Vls[buf][0],
                Vb + kb2 * 64,
                64, 128, 1280, 128, 1280, 4u);
  };
#endif

  // resident Q fragments (A layout), reused for all 20 key blocks
  bf16x16 aq[4];
  for (int kk = 0; kk < 4; ++kk) {
    const bf16* p = Qb + (size_t)(qrow0 + ln) * 128 + kk * 32 + hh * 8;
    aq[kk] = join16(*(const bf16x8*)p, *(const bf16x8*)(p + 16));
  }

  f32x8 o[8];
  for (int dt = 0; dt < 8; ++dt) o[dt] = zero8();
  float mr[8], lr[8];
  for (int r = 0; r < 8; ++r) { mr[r] = -3.0e38f; lr[r] = 0.0f; }

  const float scale = 0.08838834764831845f;   // DK^-0.5

#if HAVE_TDM
  if (wave == 0) stage(0, 0);
#endif

  for (int kb = 0; kb < 20; ++kb) {
    const int buf  = kb & 1;
    const int key0 = kb * 64;

#if HAVE_TDM
    if (wave == 0) __builtin_amdgcn_s_wait_tensorcnt(0);
    __syncthreads();                         // data ready + prev buf consumed
    if (wave == 0 && kb + 1 < 20) stage(kb + 1, buf ^ 1);
#else
    __syncthreads();                         // prev readers of buf done
    for (int i = 0; i < 8; ++i) {            // cooperative K staging
      int lin = i * 128 + tid;
      int row = lin >> 4, col = (lin & 15) * 8;
      *(bf16x8*)&Kls[buf][row * 136 + col] =
          *(const bf16x8*)(Kb + (size_t)(key0 + row) * 128 + col);
    }
    for (int i = 0; i < 8; ++i) {            // cooperative V staging
      int lin = i * 128 + tid;
      int row = lin >> 3, col = (lin & 7) * 8;
      *(bf16x8*)&Vls[buf][row * 72 + col] =
          *(const bf16x8*)(Vb + (size_t)row * 1280 + key0 + col);
    }
    __syncthreads();
#endif

    // S = Q K^T over dk=128, B-fragments from padded LDS rows
    f32x8 s[4];
    for (int nt = 0; nt < 4; ++nt) s[nt] = zero8();
    const bf16* Kbuf = &Kls[buf][0];
    for (int kk = 0; kk < 4; ++kk)
      for (int nt = 0; nt < 4; ++nt) {
        const bf16* p = Kbuf + (nt * 16 + ln) * 136 + kk * 32 + hh * 8;
        bf16x16 bk = join16(*(const bf16x8*)p, *(const bf16x8*)(p + 16));
        s[nt] = WMMA_BF16(aq[kk], bk, s[nt]);
      }
    for (int nt = 0; nt < 4; ++nt)
      for (int r = 0; r < 8; ++r) s[nt][r] *= scale;

    // online softmax: row r lives in lanes [hh*16, hh*16+16), vgpr r
    for (int r = 0; r < 8; ++r) {
      float v = fmaxf(fmaxf(s[0][r], s[1][r]), fmaxf(s[2][r], s[3][r]));
      v = fmaxf(v, __shfl_xor(v, 1));
      v = fmaxf(v, __shfl_xor(v, 2));
      v = fmaxf(v, __shfl_xor(v, 4));
      v = fmaxf(v, __shfl_xor(v, 8));
      float mn    = fmaxf(mr[r], v);
      float alpha = __expf(mr[r] - mn);
      mr[r] = mn;
      lr[r] *= alpha;
      for (int dt = 0; dt < 8; ++dt) o[dt][r] *= alpha;
      float rs = 0.0f;
      for (int nt = 0; nt < 4; ++nt) {
        float e = __expf(s[nt][r] - mn);
        s[nt][r] = e;                        // reuse S storage for P
        rs += e;
      }
      rs += __shfl_xor(rs, 1);
      rs += __shfl_xor(rs, 2);
      rs += __shfl_xor(rs, 4);
      rs += __shfl_xor(rs, 8);
      lr[r] += rs;
    }

    // P (D-layout) -> LDS [row][key] bf16, reload as A fragments
    for (int nt = 0; nt < 4; ++nt)
      for (int r = 0; r < 8; ++r)
        Pw[(hh * 8 + r) * 80 + nt * 16 + ln] = (bf16)s[nt][r];
    __asm__ volatile("s_wait_dscnt 0x0" ::: "memory");

    // O += P V over key=64
    const bf16* Vbuf = &Vls[buf][0];
    for (int kk = 0; kk < 2; ++kk) {
      const bf16* ap = Pw + ln * 80 + kk * 32 + hh * 8;
      bf16x16 pa = join16(*(const bf16x8*)ap, *(const bf16x8*)(ap + 16));
      for (int dt = 0; dt < 8; ++dt) {
        const bf16* vp = Vbuf + (dt * 16 + ln) * 72 + kk * 32 + hh * 8;
        bf16x16 bv = join16(*(const bf16x8*)vp, *(const bf16x8*)(vp + 16));
        o[dt] = WMMA_BF16(pa, bv, o[dt]);
      }
    }
  }

  // epilogue: normalize, squared distance vs query_v, reduce, atomic
  float sum = 0.0f;
  const float* qvb = QV + (size_t)b * 256 * 128;
  for (int r = 0; r < 8; ++r) {
    float inv = 1.0f / lr[r];
    int row = qrow0 + hh * 8 + r;
    for (int dt = 0; dt < 8; ++dt) {
      float ov = o[dt][r] * inv;
      float qv = qvb[(size_t)row * 128 + dt * 16 + ln];
      float d = qv - ov;
      sum += d * d;
    }
  }
  sum += __shfl_xor(sum, 1);
  sum += __shfl_xor(sum, 2);
  sum += __shfl_xor(sum, 4);
  sum += __shfl_xor(sum, 8);
  sum += __shfl_xor(sum, 16);
  if (lane == 0) atomicAdd(out + bn, -sum * (1.0f / 256.0f));
}

// ---------------------------------------------------------------------------
extern "C" void kernel_launch(void* const* d_in, const int* in_sizes, int n_in,
                              void* d_out, int out_size, void* d_ws, size_t ws_size,
                              hipStream_t stream) {
  (void)in_sizes; (void)n_in; (void)ws_size;

  const float* qrep = (const float*)d_in[0];   // [16,512,16,16]
  const float* srep = (const float*)d_in[1];   // [16,5,5,512,16,16]
  const float* wqk  = (const float*)d_in[2];   // [128,512]
  const float* wv   = (const float*)d_in[3];   // [128,512]
  float* out = (float*)d_out;                  // [16,5]

  // workspace: Qt 1MB | QV 2MB | Kt 25MB | Vn 25MB  (~53 MB, L2-resident)
  char* ws = (char*)d_ws;
  bf16*  Qt = (bf16*)ws;                                       // [16][256][128] bf16
  float* QV = (float*)(ws + (size_t)(1 << 20));                // [16][256][128] f32
  bf16*  Kt = (bf16*)(ws + (size_t)3 * (1 << 20));             // [80][1280][128] bf16
  bf16*  Vn = (bf16*)(ws + (size_t)3 * (1 << 20) + 26214400u); // [80][128][1280] bf16

  hipMemsetAsync(d_out, 0, (size_t)out_size * sizeof(float), stream);
  proj_kernel<<<dim3(832), dim3(256), 0, stream>>>(qrep, srep, wqk, wv, Qt, QV, Kt, Vn);
  attn_kernel<<<dim3(320), dim3(128), 0, stream>>>(Qt, QV, Kt, Vn, out);
}
